// SwinTransformerBlock_15118284882065
// MI455X (gfx1250) — compile-verified, matrix operand order checked by reading, not run on hardware
//
#include <hip/hip_runtime.h>
#include <hip/hip_bf16.h>
#include <cstdint>

// ---------------------------------------------------------------------------
// Types for CDNA5 WMMA (wave32)
// ---------------------------------------------------------------------------
typedef __attribute__((ext_vector_type(16))) _Float16 v16h;
typedef __attribute__((ext_vector_type(8)))  _Float16 v8h;
typedef __attribute__((ext_vector_type(8)))  float    v8f;

__device__ __forceinline__ v16h combine16(v8h lo, v8h hi) {
    v16h a;
#pragma unroll
    for (int i = 0; i < 8; ++i) { a[i] = lo[i]; a[i + 8] = hi[i]; }
    return a;
}

__device__ __forceinline__ v8f wmma_f16(v16h a, v16h b, v8f c) {
    // D = A(16x32 f16) x B(32x16 f16) + C(16x16 f32)
    return __builtin_amdgcn_wmma_f32_16x16x32_f16(false, a, false, b,
                                                  (short)0, c, false, false);
}

// ---------------------------------------------------------------------------
// Problem constants (B=32, H=W=64, C=192, heads=6, d=32, win=8, shift=4)
// ---------------------------------------------------------------------------
#define TOK      131072     // B * H * W
#define CDIM     192
#define HID      768
#define QKVDIM   576
#define NWIN     2048       // total windows (B * 64)
#define QSCALE   0.17677669529663687f   // 32^-0.5

__device__ __forceinline__ int shift_region(int p) {
    // regions from slices (0,-8), (-8,-4), (-4,None) on 64
    return p < 56 ? 0 : (p < 60 ? 1 : 2);
}

// ---------------------------------------------------------------------------
// f32 -> f16 weight conversion
// ---------------------------------------------------------------------------
__global__ void cvt_f16_kernel(const float* __restrict__ src,
                               _Float16* __restrict__ dst, int n) {
    int i = blockIdx.x * 256 + threadIdx.x;
    if (i < n) dst[i] = (_Float16)src[i];
}

// ---------------------------------------------------------------------------
// LayerNorm (one wave of 32 lanes per token, C=192 -> 6 values/lane).
// GATHER=1: destination is windowed order (roll -4,-4 + 8x8 window partition),
//           source is natural token order (used for LN1).
// GATHER=0: identity indexing (used for LN2).
// ---------------------------------------------------------------------------
template <int GATHER>
__global__ __launch_bounds__(256)
void ln_kernel(const float* __restrict__ xin, const float* __restrict__ gam,
               const float* __restrict__ bet, _Float16* __restrict__ out) {
    const int lane = threadIdx.x & 31;
    const int t = blockIdx.x * 8 + (threadIdx.x >> 5);   // token (dest order)

    size_t src;
    if (GATHER) {
        int b = t >> 12, rem = t & 4095;
        int wv = rem >> 6, n = rem & 63;
        int wh = wv >> 3, ww = wv & 7, i = n >> 3, j = n & 7;
        int r = wh * 8 + i, c = ww * 8 + j;
        int sr = (r + 4) & 63, sc = (c + 4) & 63;         // roll(-4) gather
        src = ((size_t)(b << 12) + (sr << 6) + sc) * CDIM;
    } else {
        src = (size_t)t * CDIM;
    }

    float v[6];
#pragma unroll
    for (int k = 0; k < 6; ++k) v[k] = xin[src + lane + 32 * k];

    float s = 0.f;
#pragma unroll
    for (int k = 0; k < 6; ++k) s += v[k];
#pragma unroll
    for (int off = 16; off > 0; off >>= 1) s += __shfl_xor(s, off);
    float mu = s * (1.0f / CDIM);

    float q = 0.f;
#pragma unroll
    for (int k = 0; k < 6; ++k) { float d = v[k] - mu; q += d * d; }
#pragma unroll
    for (int off = 16; off > 0; off >>= 1) q += __shfl_xor(q, off);
    float rs = rsqrtf(q * (1.0f / CDIM) + 1e-5f);

#pragma unroll
    for (int k = 0; k < 6; ++k) {
        int c = lane + 32 * k;
        out[(size_t)t * CDIM + c] = (_Float16)((v[k] - mu) * rs * gam[c] + bet[c]);
    }
}

// ---------------------------------------------------------------------------
// Generic WMMA GEMM:  Out = A[M x K](f16) * W^T  (+bias, epilogue by MODE)
//   W is [N x K] f16 row-major (so B-matrix[k][n] = W[n*K + k]).
//   Block tile: 128 rows x 64 cols, 256 threads = 8 waves.
//   Wave w owns row-strip mt = w (16 rows) and all 4 N-tiles:
//   per K-step 1 A-fragment feeds 4 WMMAs.
//   A-tile staged in LDS via GLOBAL_LOAD_ASYNC_TO_LDS_B128 (ASYNCcnt path).
// MODE 0: QKV   -> +bias, scale cols<192 by 32^-0.5, f16 out (stride 576)
// MODE 1: PROJ  -> +bias, un-window/un-roll scatter, +shortcut, f32 out
// MODE 2: FC1   -> +bias, exact GELU, f16 out (stride 768)
// MODE 3: FC2   -> +bias, accumulate into f32 out (stride 192)
// ---------------------------------------------------------------------------
template <int MODE>
__global__ __launch_bounds__(256)
void gemm_ep_kernel(const _Float16* __restrict__ A, const _Float16* __restrict__ Wt,
                    const float* __restrict__ bias, _Float16* __restrict__ outh,
                    float* __restrict__ outf, const float* __restrict__ xin,
                    int K) {
    __shared__ __align__(16) _Float16 As[128][192];

    const int tid = threadIdx.x;
    const int wave = tid >> 5, lane = tid & 31;
    const int lane15 = lane & 15, g = lane >> 4;
    const int kbase = g * 8;
    const int m0 = blockIdx.x * 128;
    const int n0 = blockIdx.y * 64;
    const int mt = wave;                       // one 16-row strip per wave

    // pull the (L2-resident) weight tile toward this WGP
    __builtin_prefetch(Wt + (size_t)(n0 + (tid & 63)) * K, 0, 1);

    v8f acc[4];
#pragma unroll
    for (int i = 0; i < 4; ++i) acc[i] = {};

    const int nchunk = K / 192;
    for (int kc = 0; kc < nchunk; ++kc) {
        __syncthreads();   // previous chunk's LDS reads done before overwrite
        // stage 128x192 f16 A-chunk: 3072 x 16B async direct-to-LDS copies
        const unsigned long long gbase =
            (unsigned long long)(uintptr_t)(A + (size_t)m0 * K + kc * 192);
#pragma unroll
        for (int it = 0; it < 12; ++it) {
            int idx = tid + it * 256;
            int row = idx / 24, col16 = idx % 24;
            unsigned int ldsb = (unsigned int)(uintptr_t)(&As[row][col16 * 8]);
            unsigned int goff = (unsigned int)(row * K + col16 * 8) * 2u;
            asm volatile("global_load_async_to_lds_b128 %0, %1, %2"
                         :: "v"(ldsb), "v"(goff), "s"(gbase)
                         : "memory");
        }
        asm volatile("s_wait_asynccnt 0" ::: "memory");
        __syncthreads();

#pragma unroll
        for (int ks = 0; ks < 6; ++ks) {
            const _Float16* ap = &As[mt * 16 + lane15][ks * 32 + kbase];
            v16h af = combine16(*(const v8h*)ap, *(const v8h*)(ap + 16));
#pragma unroll
            for (int nt = 0; nt < 4; ++nt) {
                int n = n0 + nt * 16 + lane15;
                const _Float16* wp = Wt + (size_t)n * K + kc * 192 + ks * 32 + kbase;
                v16h bf = combine16(*(const v8h*)wp, *(const v8h*)(wp + 16));
                acc[nt] = wmma_f16(af, bf, acc[nt]);
            }
        }
    }

    // epilogue: element (row, cn) = acc[nt][r] at lane (cn-local), half g
#pragma unroll
    for (int nt = 0; nt < 4; ++nt) {
        int cn = n0 + nt * 16 + lane15;
        float bval = bias[cn];
#pragma unroll
        for (int r = 0; r < 8; ++r) {
            int row = m0 + mt * 16 + r + 8 * g;
            float val = acc[nt][r] + bval;
            if (MODE == 0) {
                if (cn < 192) val *= QSCALE;          // pre-scale q
                outh[(size_t)row * QKVDIM + cn] = (_Float16)val;
            } else if (MODE == 1) {
                int b = row >> 12, rem = row & 4095;
                int wv = rem >> 6, nn = rem & 63;
                int wh = wv >> 3, ww = wv & 7, i = nn >> 3, j = nn & 7;
                int rr = wh * 8 + i, cc = ww * 8 + j;
                int sr = (rr + 4) & 63, sc = (cc + 4) & 63; // roll(+4) scatter
                size_t gidx = ((size_t)(b << 12) + (sr << 6) + sc) * CDIM + cn;
                outf[gidx] = xin[gidx] + val;         // shortcut + proj
            } else if (MODE == 2) {
                float ge = 0.5f * val * (1.0f + erff(val * 0.70710678118654752f));
                outh[(size_t)row * HID + cn] = (_Float16)ge;
            } else {
                outf[(size_t)row * CDIM + cn] += val; // x2 + mlp
            }
        }
    }
}

// ---------------------------------------------------------------------------
// Windowed attention: one block per window (2048), 6 waves = 6 heads.
//   S = q k^T (+rpe +shift-mask) -> softmax -> P (f16 in per-wave LDS)
//   out = P v, written as f16 [token][C] (C index = head*32 + d)
// ---------------------------------------------------------------------------
__global__ __launch_bounds__(192)
void attn_kernel(const _Float16* __restrict__ qkv, const float* __restrict__ rpb,
                 _Float16* __restrict__ attnout) {
    __shared__ __align__(16) float     s_rpb[225 * 6];
    __shared__ __align__(16) _Float16 s_P[6][64][72];

    const int tid = threadIdx.x;
    for (int i = tid; i < 225 * 6; i += 192) s_rpb[i] = rpb[i];
    __syncthreads();

    const int hd = tid >> 5;                    // head = wave
    const int lane = tid & 31, lane15 = lane & 15, g = lane >> 4;
    const int kbase = g * 8;
    const int wv = blockIdx.x;
    const int wvl = wv & 63, wh = wvl >> 3, ww = wvl & 7;
    const size_t base = (size_t)wv * 64 * QKVDIM;

    // ---- S = q k^T -------------------------------------------------------
    v16h aq[4];
#pragma unroll
    for (int mtq = 0; mtq < 4; ++mtq) {
        const _Float16* qp = qkv + base + (size_t)(mtq * 16 + lane15) * QKVDIM
                           + hd * 32 + kbase;
        aq[mtq] = combine16(*(const v8h*)qp, *(const v8h*)(qp + 16));
    }
    v8f acc[4][4];
#pragma unroll
    for (int nt = 0; nt < 4; ++nt) {
        const _Float16* kp = qkv + base + (size_t)(nt * 16 + lane15) * QKVDIM
                           + 192 + hd * 32 + kbase;
        v16h bk = combine16(*(const v8h*)kp, *(const v8h*)(kp + 16));
#pragma unroll
        for (int mtq = 0; mtq < 4; ++mtq) {
            v8f z = {};
            acc[mtq][nt] = wmma_f16(aq[mtq], bk, z);
        }
    }

    // ---- key-side constants for this lane's 4 columns --------------------
    int i2a[4], j2a[4], labn[4];
#pragma unroll
    for (int nt = 0; nt < 4; ++nt) {
        int n = nt * 16 + lane15;
        i2a[nt] = n >> 3; j2a[nt] = n & 7;
        labn[nt] = shift_region(wh * 8 + i2a[nt]) * 3 +
                   shift_region(ww * 8 + j2a[nt]);
    }

    // ---- +rpe +mask, row softmax (rows live in 16-lane halves) -----------
#pragma unroll
    for (int mtq = 0; mtq < 4; ++mtq) {
#pragma unroll
        for (int r = 0; r < 8; ++r) {
            int m = mtq * 16 + r + 8 * g;
            int i1 = m >> 3, j1 = m & 7;
            int labm = shift_region(wh * 8 + i1) * 3 +
                       shift_region(ww * 8 + j1);
            float v[4];
#pragma unroll
            for (int nt = 0; nt < 4; ++nt) {
                float rp = s_rpb[((i1 - i2a[nt] + 7) * 15 + (j1 - j2a[nt] + 7)) * 6 + hd];
                v[nt] = acc[mtq][nt][r] + rp + (labm == labn[nt] ? 0.f : -100.f);
            }
            float mx = fmaxf(fmaxf(v[0], v[1]), fmaxf(v[2], v[3]));
            mx = fmaxf(mx, __shfl_xor(mx, 1));
            mx = fmaxf(mx, __shfl_xor(mx, 2));
            mx = fmaxf(mx, __shfl_xor(mx, 4));
            mx = fmaxf(mx, __shfl_xor(mx, 8));
            float e[4], s = 0.f;
#pragma unroll
            for (int nt = 0; nt < 4; ++nt) { e[nt] = __expf(v[nt] - mx); s += e[nt]; }
            s += __shfl_xor(s, 1);
            s += __shfl_xor(s, 2);
            s += __shfl_xor(s, 4);
            s += __shfl_xor(s, 8);
            float inv = 1.0f / s;
#pragma unroll
            for (int nt = 0; nt < 4; ++nt)
                s_P[hd][m][nt * 16 + lane15] = (_Float16)(e[nt] * inv);
        }
    }

    // ---- out = P v  (K=64 -> 2 steps, N=32 -> 2 tiles) -------------------
    v8f accO[4][2];
#pragma unroll
    for (int mtq = 0; mtq < 4; ++mtq) { accO[mtq][0] = {}; accO[mtq][1] = {}; }

#pragma unroll
    for (int ks = 0; ks < 2; ++ks) {
        v16h ap[4];
#pragma unroll
        for (int mtq = 0; mtq < 4; ++mtq) {
            const _Float16* pp = &s_P[hd][mtq * 16 + lane15][ks * 32 + kbase];
            ap[mtq] = combine16(*(const v8h*)pp, *(const v8h*)(pp + 16));
        }
#pragma unroll
        for (int nt = 0; nt < 2; ++nt) {
            int dd = nt * 16 + lane15;
            const _Float16* vb = qkv + base + 384 + hd * 32 + dd
                               + (size_t)(ks * 32 + kbase) * QKVDIM;
            v16h bv;
#pragma unroll
            for (int h = 0; h < 8; ++h) {
                bv[h]     = vb[(size_t)h * QKVDIM];
                bv[h + 8] = vb[(size_t)(h + 16) * QKVDIM];
            }
#pragma unroll
            for (int mtq = 0; mtq < 4; ++mtq)
                accO[mtq][nt] = wmma_f16(ap[mtq], bv, accO[mtq][nt]);
        }
    }

#pragma unroll
    for (int mtq = 0; mtq < 4; ++mtq)
#pragma unroll
        for (int nt = 0; nt < 2; ++nt)
#pragma unroll
            for (int r = 0; r < 8; ++r) {
                int m = mtq * 16 + r + 8 * g;
                int dd = nt * 16 + lane15;
                attnout[((size_t)wv * 64 + m) * CDIM + hd * 32 + dd] =
                    (_Float16)accO[mtq][nt][r];
            }
}

// ---------------------------------------------------------------------------
// Host-side launcher
// ---------------------------------------------------------------------------
extern "C" void kernel_launch(void* const* d_in, const int* in_sizes, int n_in,
                              void* d_out, int out_size, void* d_ws, size_t ws_size,
                              hipStream_t stream) {
    (void)in_sizes; (void)n_in; (void)out_size; (void)ws_size;

    const float* x       = (const float*)d_in[0];
    const float* n1g     = (const float*)d_in[1];
    const float* n1b     = (const float*)d_in[2];
    const float* qkv_w   = (const float*)d_in[3];
    const float* qkv_b   = (const float*)d_in[4];
    const float* rpb     = (const float*)d_in[5];
    const float* proj_w  = (const float*)d_in[6];
    const float* proj_b  = (const float*)d_in[7];
    const float* n2g     = (const float*)d_in[8];
    const float* n2b     = (const float*)d_in[9];
    const float* fc1_w   = (const float*)d_in[10];
    const float* fc1_b   = (const float*)d_in[11];
    const float* fc2_w   = (const float*)d_in[12];
    const float* fc2_b   = (const float*)d_in[13];
    float* out = (float*)d_out;

    // workspace layout (bytes)
    char* ws = (char*)d_ws;
    _Float16* W16Q  = (_Float16*)(ws + 0);                 // 576*192
    _Float16* W16P  = (_Float16*)(ws + 221184);            // 192*192
    _Float16* W16F1 = (_Float16*)(ws + 294912);            // 768*192
    _Float16* W16F2 = (_Float16*)(ws + 589824);            // 192*768
    _Float16* YW    = (_Float16*)(ws + 1048576);           // 131072*192 f16
    _Float16* QKV   = (_Float16*)(ws + 51380224);          // 131072*576 f16
    _Float16* ATTN  = (_Float16*)(ws + 202375168);         // 131072*192 f16
    _Float16* LN2   = YW;                                  // reuse (YW dead)
    _Float16* HBUF  = QKV;                                 // reuse QKV+ATTN region

    // 1. weights -> f16
    cvt_f16_kernel<<<(110592 + 255) / 256, 256, 0, stream>>>(qkv_w,  W16Q, 110592);
    cvt_f16_kernel<<<(36864  + 255) / 256, 256, 0, stream>>>(proj_w, W16P, 36864);
    cvt_f16_kernel<<<(147456 + 255) / 256, 256, 0, stream>>>(fc1_w,  W16F1, 147456);
    cvt_f16_kernel<<<(147456 + 255) / 256, 256, 0, stream>>>(fc2_w,  W16F2, 147456);

    // 2. LN1 + roll + window-partition -> f16
    ln_kernel<1><<<TOK / 8, 256, 0, stream>>>(x, n1g, n1b, YW);

    // 3. QKV gemm (q pre-scaled)
    gemm_ep_kernel<0><<<dim3(TOK / 128, QKVDIM / 64), 256, 0, stream>>>(
        YW, W16Q, qkv_b, QKV, nullptr, nullptr, 192);

    // 4. windowed attention
    attn_kernel<<<NWIN, 192, 0, stream>>>(QKV, rpb, ATTN);

    // 5. proj + un-roll scatter + shortcut  -> d_out = x2
    gemm_ep_kernel<1><<<dim3(TOK / 128, CDIM / 64), 256, 0, stream>>>(
        ATTN, W16P, proj_b, nullptr, out, x, 192);

    // 6. LN2 on x2
    ln_kernel<0><<<TOK / 8, 256, 0, stream>>>(out, n2g, n2b, LN2);

    // 7. FC1 + GELU
    gemm_ep_kernel<2><<<dim3(TOK / 128, HID / 64), 256, 0, stream>>>(
        LN2, W16F1, fc1_b, HBUF, nullptr, nullptr, 192);

    // 8. FC2, accumulate into d_out
    gemm_ep_kernel<3><<<dim3(TOK / 128, CDIM / 64), 256, 0, stream>>>(
        HBUF, W16F2, fc2_b, nullptr, out, nullptr, 768);
}
